// GeneralSelfAttention_66511863545907
// MI455X (gfx1250) — matrix operand order repaired and checked by reading, not compiled, not run on hardware
//
#include <hip/hip_runtime.h>
#include <hip/hip_bf16.h>

// ---------------------------------------------------------------------------
// Types for CDNA5 WMMA (wave32): v_wmma_f32_16x16x32_bf16
// ---------------------------------------------------------------------------
typedef __bf16 v16bf __attribute__((ext_vector_type(16)));
typedef __bf16 v8bf  __attribute__((ext_vector_type(8)));
typedef float  v8f   __attribute__((ext_vector_type(8)));

#define LOG2E 1.4426950408889634f

static __device__ inline v8f vzero8() {
  v8f z;
#pragma unroll
  for (int i = 0; i < 8; ++i) z[i] = 0.0f;
  return z;
}

static __device__ inline v16bf combine16(v8bf a, v8bf b) {
  v16bf r;
#pragma unroll
  for (int i = 0; i < 8; ++i) { r[i] = a[i]; r[i + 8] = b[i]; }
  return r;
}

// A-matrix fragment (16x32 bf16, row-major source).
// Lane l holds row (base_row + l%16).  Elements:
//   e in [0,8):  K = kk + 8*(l>=16) + e        (contiguous 16B)
//   e in [8,16): K = kk + 16 + 8*(l>=16) + e-8 (contiguous 16B)
static __device__ inline v16bf load_a_frag(const __bf16* __restrict__ base,
                                           int row, int ld, int kk,
                                           int l16, int lhalf) {
  const __bf16* p = base + (size_t)(row + l16) * ld + kk + lhalf * 8;
  v8bf a0 = *(const v8bf*)(p);
  v8bf a1 = *(const v8bf*)(p + 16);
  return combine16(a0, a1);
}

// B-matrix fragment (32x16 bf16) from an N-major source Bt[n, k] (ld = ldk).
// Lane l holds column (col + l%16).  Elements:
//   e in [0,8):  K = kk + 8*(l>=16) + e        (contiguous 16B)
//   e in [8,16): K = kk + 16 + 8*(l>=16) + e-8 (contiguous 16B)
static __device__ inline v16bf load_b_frag(const __bf16* __restrict__ bt,
                                           int col, int ldk, int kk,
                                           int l16, int lhalf) {
  const __bf16* p = bt + (size_t)(col + l16) * ldk + kk + lhalf * 8;
  v8bf b0 = *(const v8bf*)(p);
  v8bf b1 = *(const v8bf*)(p + 16);
  return combine16(b0, b1);
}

static __device__ inline v8f wmma_bf16(v16bf a, v16bf b, v8f c) {
  return __builtin_amdgcn_wmma_f32_16x16x32_bf16(
      /*neg_a=*/false, a, /*neg_b=*/false, b,
      /*c_mod=*/(short)0, c, /*reuse_a=*/false, /*reuse_b=*/false);
}

// ---------------------------------------------------------------------------
// Problem constants
// ---------------------------------------------------------------------------
#define BB   2
#define SS   2048
#define DD   1024
#define HH   16
#define DKK  64
#define MM   (BB * SS)   // 4096 rows

// ---------------------------------------------------------------------------
// Elementwise fp32 -> bf16 convert
// ---------------------------------------------------------------------------
__global__ void cvt_bf16_kernel(const float* __restrict__ in,
                                __bf16* __restrict__ out, int n) {
  for (int i = blockIdx.x * blockDim.x + threadIdx.x; i < n;
       i += gridDim.x * blockDim.x)
    out[i] = (__bf16)in[i];
}

// Batched transpose + convert: in[m][r][c] (f32) -> out[m][c][r] (bf16)
__global__ void transpose_bf16_kernel(const float* __restrict__ in,
                                      __bf16* __restrict__ out,
                                      int R, int C, int n) {
  for (int i = blockIdx.x * blockDim.x + threadIdx.x; i < n;
       i += gridDim.x * blockDim.x) {
    int m = i / (R * C);
    int rem = i - m * (R * C);
    int r = rem / C;
    int c = rem - r * C;
    out[(size_t)m * R * C + (size_t)c * R + r] = (__bf16)in[i];
  }
}

// ---------------------------------------------------------------------------
// QKV projection: per head h, z in {Q,K,V}:
//   P[b,s,:] = x[b,s,:] @ W[h] + bias[h]           (K = 1024, N = 64)
// Q gets  log2(e)/(tau*sqrt(dk))  folded in (softmax later uses exp2).
// Q,K written row-major [b,h,s,dk]; V written transposed [b,h,dk,s].
// Block: 256 thr = 8 waves; wave computes a 32x64 slab (2 row tiles sharing
// every B fragment -> 8 WMMAs per 12 b128 loads).  Block = 256 rows.
// ---------------------------------------------------------------------------
__global__ __launch_bounds__(256) void proj_qkv_kernel(
    const __bf16* __restrict__ xb,                       // [M, D]
    const __bf16* __restrict__ WqT, const __bf16* __restrict__ WkT,
    const __bf16* __restrict__ WvT,                      // each [H][DK][D]
    const float* __restrict__ bq, const float* __restrict__ bk,
    const float* __restrict__ bv, const float* __restrict__ tau,
    __bf16* __restrict__ Qb, __bf16* __restrict__ Kb,    // [B,H,S,DK]
    __bf16* __restrict__ Vt)                             // [B,H,DK,S]
{
  const int w     = threadIdx.x >> 5;
  const int lane  = threadIdx.x & 31;
  const int l16   = lane & 15;
  const int lhalf = lane >> 4;
  const int rbase = blockIdx.x * 256 + w * 32;
  const int h     = blockIdx.y;
  const int z     = blockIdx.z;

  const __bf16* Bt;
  const float*  bias;
  if (z == 0)      { Bt = WqT; bias = bq; }
  else if (z == 1) { Bt = WkT; bias = bk; }
  else             { Bt = WvT; bias = bv; }
  Bt += (size_t)h * DKK * DD;

  v8f acc[2][4];
#pragma unroll
  for (int q = 0; q < 2; ++q)
#pragma unroll
    for (int t = 0; t < 4; ++t) acc[q][t] = vzero8();

  for (int kk = 0; kk < DD; kk += 32) {
    const v16bf a0 = load_a_frag(xb, rbase,      DD, kk, l16, lhalf);
    const v16bf a1 = load_a_frag(xb, rbase + 16, DD, kk, l16, lhalf);
#pragma unroll
    for (int t = 0; t < 4; ++t) {
      const v16bf b = load_b_frag(Bt, t * 16, DD, kk, l16, lhalf);
      acc[0][t] = wmma_bf16(a0, b, acc[0][t]);
      acc[1][t] = wmma_bf16(a1, b, acc[1][t]);
    }
  }

  // sqrt(64)=8; Q also gets log2(e) so attention can use exp2 directly
  const float sc =
      (z == 0) ? (__builtin_amdgcn_rcpf(tau[h] * 8.0f) * LOG2E) : 1.0f;
#pragma unroll
  for (int q = 0; q < 2; ++q)
#pragma unroll
    for (int t = 0; t < 4; ++t) {
      const int n = t * 16 + l16;
      const float bcol = bias[h * DKK + n];
#pragma unroll
      for (int j = 0; j < 8; ++j) {
        const int r = rbase + q * 16 + j + lhalf * 8;   // global row in [0,M)
        const int bIdx = r >> 11;                        // r / S
        const int s    = r & (SS - 1);
        const float v = (acc[q][t][j] + bcol) * sc;
        if (z == 2) {
          Vt[((size_t)(bIdx * HH + h) * DKK + n) * SS + s] = (__bf16)v;
        } else {
          __bf16* out = (z == 0) ? Qb : Kb;
          out[((size_t)(bIdx * HH + h) * SS + s) * DKK + n] = (__bf16)v;
        }
      }
    }
}

// ---------------------------------------------------------------------------
// Flash attention: block = 128 thr = 4 waves; wave handles 32 query rows
// (two 16-row tiles sharing every K/V B fragment -> 16 WMMAs per 16 loads);
// block covers 128 queries of one (b,h).  Keys streamed in tiles of 32.
// Scores arrive pre-scaled by log2(e)/(tau*sqrt(dk)) -> exp2-domain softmax.
// ---------------------------------------------------------------------------
__global__ __launch_bounds__(128) void attn_kernel(
    const __bf16* __restrict__ Qb, const __bf16* __restrict__ Kb,
    const __bf16* __restrict__ Vt, __bf16* __restrict__ Ob)
{
  __shared__ __align__(16) __bf16 lp[4][2][16][32];   // per-wave P staging

  const int w     = threadIdx.x >> 5;
  const int lane  = threadIdx.x & 31;
  const int l16   = lane & 15;
  const int lhalf = lane >> 4;
  const int qbase = blockIdx.x * 128 + w * 32;
  const int h     = blockIdx.y;
  const int b     = blockIdx.z;

  const __bf16* Qp = Qb + (size_t)(b * HH + h) * SS * DKK;
  const __bf16* Kp = Kb + (size_t)(b * HH + h) * SS * DKK;
  const __bf16* Vp = Vt + (size_t)(b * HH + h) * DKK * SS;

  // Q fragments [qt][d-slice], resident for the whole loop.
  v16bf qa[2][2];
#pragma unroll
  for (int q = 0; q < 2; ++q) {
    qa[q][0] = load_a_frag(Qp, qbase + q * 16, DKK, 0,  l16, lhalf);
    qa[q][1] = load_a_frag(Qp, qbase + q * 16, DKK, 32, l16, lhalf);
  }

  v8f o[2][4];
  float m[2][8], l[2][8];
#pragma unroll
  for (int q = 0; q < 2; ++q) {
#pragma unroll
    for (int t = 0; t < 4; ++t) o[q][t] = vzero8();
#pragma unroll
    for (int j = 0; j < 8; ++j) { m[q][j] = -INFINITY; l[q][j] = 0.0f; }
  }

  for (int kb = 0; kb < SS; kb += 32) {
    // prefetch next key/value tiles toward the WGP cache (L2-resident stream)
    if (kb + 32 < SS) {
      __builtin_prefetch(Kp + (size_t)(kb + 32 + l16) * DKK, 0, 3);
      __builtin_prefetch(Vp + (size_t)l16 * SS + kb + 32, 0, 3);
    }

    // ---- K tile fragments, shared by both query tiles ----
    const v16bf kb00 = load_b_frag(Kp, kb,      DKK, 0,  l16, lhalf);
    const v16bf kb01 = load_b_frag(Kp, kb,      DKK, 32, l16, lhalf);
    const v16bf kb10 = load_b_frag(Kp, kb + 16, DKK, 0,  l16, lhalf);
    const v16bf kb11 = load_b_frag(Kp, kb + 16, DKK, 32, l16, lhalf);

    v8f s[2][2];
#pragma unroll
    for (int q = 0; q < 2; ++q) {
      s[q][0] = vzero8();
      s[q][0] = wmma_bf16(qa[q][0], kb00, s[q][0]);
      s[q][0] = wmma_bf16(qa[q][1], kb01, s[q][0]);
      s[q][1] = vzero8();
      s[q][1] = wmma_bf16(qa[q][0], kb10, s[q][1]);
      s[q][1] = wmma_bf16(qa[q][1], kb11, s[q][1]);
    }

    // ---- online softmax per query tile (exp2 domain) ----
#pragma unroll
    for (int q = 0; q < 2; ++q) {
      float tm[8];
#pragma unroll
      for (int j = 0; j < 8; ++j) tm[j] = fmaxf(s[q][0][j], s[q][1][j]);
#pragma unroll
      for (int off = 1; off < 16; off <<= 1)
#pragma unroll
        for (int j = 0; j < 8; ++j)
          tm[j] = fmaxf(tm[j], __shfl_xor(tm[j], off, 32));

      float alpha[8];
#pragma unroll
      for (int j = 0; j < 8; ++j) {
        const float mn = fmaxf(m[q][j], tm[j]);
        alpha[j] = exp2f(m[q][j] - mn);
        m[q][j] = mn;
      }
      float p0[8], p1[8], rs[8];
#pragma unroll
      for (int j = 0; j < 8; ++j) {
        p0[j] = exp2f(s[q][0][j] - m[q][j]);
        p1[j] = exp2f(s[q][1][j] - m[q][j]);
        rs[j] = p0[j] + p1[j];
      }
#pragma unroll
      for (int off = 1; off < 16; off <<= 1)
#pragma unroll
        for (int j = 0; j < 8; ++j)
          rs[j] += __shfl_xor(rs[j], off, 32);
#pragma unroll
      for (int j = 0; j < 8; ++j) l[q][j] = l[q][j] * alpha[j] + rs[j];
#pragma unroll
      for (int t = 0; t < 4; ++t)
#pragma unroll
        for (int j = 0; j < 8; ++j) o[q][t][j] *= alpha[j];

      // C-layout f32 -> bf16 staging for the A-layout reload
#pragma unroll
      for (int j = 0; j < 8; ++j) {
        const int r = j + lhalf * 8;
        lp[w][q][r][l16]      = (__bf16)p0[j];
        lp[w][q][r][l16 + 16] = (__bf16)p1[j];
      }
    }

    // intra-wave DS ordering only; wait for the P stores to land
    asm volatile("s_wait_dscnt 0" ::: "memory");

    v16bf pa[2];
#pragma unroll
    for (int q = 0; q < 2; ++q) {
      const __bf16* pr = &lp[w][q][l16][lhalf * 8];
      v8bf x0 = *(const v8bf*)(pr);
      v8bf x1 = *(const v8bf*)(pr + 16);
      pa[q] = combine16(x0, x1);
    }

    // ---- O += P @ V; V fragments shared by both query tiles ----
#pragma unroll
    for (int t = 0; t < 4; ++t) {
      const v16bf vb = load_b_frag(Vp, t * 16, SS, kb, l16, lhalf);
      o[0][t] = wmma_bf16(pa[0], vb, o[0][t]);
      o[1][t] = wmma_bf16(pa[1], vb, o[1][t]);
    }
  }

  // ---- normalize and write concat-heads bf16 output ----
#pragma unroll
  for (int q = 0; q < 2; ++q) {
    float inv[8];
#pragma unroll
    for (int j = 0; j < 8; ++j) inv[j] = __builtin_amdgcn_rcpf(l[q][j]);
#pragma unroll
    for (int t = 0; t < 4; ++t)
#pragma unroll
      for (int j = 0; j < 8; ++j) {
        const int srow = qbase + q * 16 + j + lhalf * 8;
        Ob[((size_t)b * SS + srow) * DD + h * DKK + t * 16 + l16] =
            (__bf16)(o[q][t][j] * inv[j]);
      }
  }
}

// ---------------------------------------------------------------------------
// Output projection: out[M, D] = Ob[M, D] @ Wo[D, D] + bo  (fp32 out)
// Wave computes a 32x64 slab (2 row tiles share every B fragment).
// ---------------------------------------------------------------------------
__global__ __launch_bounds__(256) void outproj_kernel(
    const __bf16* __restrict__ Ob, const __bf16* __restrict__ WoT,
    const float* __restrict__ bo, float* __restrict__ out)
{
  const int w     = threadIdx.x >> 5;
  const int lane  = threadIdx.x & 31;
  const int l16   = lane & 15;
  const int lhalf = lane >> 4;
  const int rbase = blockIdx.x * 256 + w * 32;
  const int cbase = blockIdx.y * 64;

  v8f acc[2][4];
#pragma unroll
  for (int q = 0; q < 2; ++q)
#pragma unroll
    for (int t = 0; t < 4; ++t) acc[q][t] = vzero8();

  for (int kk = 0; kk < DD; kk += 32) {
    const v16bf a0 = load_a_frag(Ob, rbase,      DD, kk, l16, lhalf);
    const v16bf a1 = load_a_frag(Ob, rbase + 16, DD, kk, l16, lhalf);
#pragma unroll
    for (int t = 0; t < 4; ++t) {
      const v16bf b = load_b_frag(WoT, cbase + t * 16, DD, kk, l16, lhalf);
      acc[0][t] = wmma_bf16(a0, b, acc[0][t]);
      acc[1][t] = wmma_bf16(a1, b, acc[1][t]);
    }
  }

#pragma unroll
  for (int q = 0; q < 2; ++q)
#pragma unroll
    for (int t = 0; t < 4; ++t) {
      const int n = cbase + t * 16 + l16;
      const float bc = bo[n];
#pragma unroll
      for (int j = 0; j < 8; ++j) {
        const int r = rbase + q * 16 + j + lhalf * 8;
        out[(size_t)r * DD + n] = acc[q][t][j] + bc;
      }
    }
}

// ---------------------------------------------------------------------------
// Host entry
// ---------------------------------------------------------------------------
extern "C" void kernel_launch(void* const* d_in, const int* in_sizes, int n_in,
                              void* d_out, int out_size, void* d_ws,
                              size_t ws_size, hipStream_t stream) {
  const float* x   = (const float*)d_in[0];
  const float* Wq  = (const float*)d_in[1];
  const float* bq  = (const float*)d_in[2];
  const float* Wk  = (const float*)d_in[3];
  const float* bk  = (const float*)d_in[4];
  const float* Wv  = (const float*)d_in[5];
  const float* bv  = (const float*)d_in[6];
  const float* tau = (const float*)d_in[7];
  const float* Wo  = (const float*)d_in[8];
  const float* bo  = (const float*)d_in[9];
  float* out = (float*)d_out;

  // Workspace layout (bytes); all 2^k-aligned.
  char* ws = (char*)d_ws;
  __bf16* xb  = (__bf16*)(ws);                       //  8 MB: [M, D]
  __bf16* WqT = (__bf16*)(ws + (8u << 20));          //  2 MB: [H][DK][D]
  __bf16* WkT = (__bf16*)(ws + (10u << 20));         //  2 MB
  __bf16* WvT = (__bf16*)(ws + (12u << 20));         //  2 MB
  __bf16* WoT = (__bf16*)(ws + (14u << 20));         //  2 MB: [D][D] (n-major)
  __bf16* Qb  = (__bf16*)(ws + (16u << 20));         //  8 MB: [B,H,S,DK]
  __bf16* Kb  = (__bf16*)(ws + (24u << 20));         //  8 MB: [B,H,S,DK]
  __bf16* Vt  = (__bf16*)(ws + (32u << 20));         //  8 MB: [B,H,DK,S]
  __bf16* Ob  = (__bf16*)(ws + (40u << 20));         //  8 MB: [B,S,H*DK]

  // 1) convert x to bf16
  cvt_bf16_kernel<<<2048, 256, 0, stream>>>(x, xb, MM * DD);

  // 2) transpose + convert weights to N-major bf16
  {
    const int nqkv = HH * DD * DKK;
    transpose_bf16_kernel<<<1024, 256, 0, stream>>>(Wq, WqT, DD, DKK, nqkv);
    transpose_bf16_kernel<<<1024, 256, 0, stream>>>(Wk, WkT, DD, DKK, nqkv);
    transpose_bf16_kernel<<<1024, 256, 0, stream>>>(Wv, WvT, DD, DKK, nqkv);
    transpose_bf16_kernel<<<1024, 256, 0, stream>>>(Wo, WoT, DD, DD, DD * DD);
  }
  // 3) QKV projections (z: 0=Q scaled, 1=K, 2=V-transposed)
  proj_qkv_kernel<<<dim3(MM / 256, HH, 3), 256, 0, stream>>>(
      xb, WqT, WkT, WvT, bq, bk, bv, tau, Qb, Kb, Vt);
  // 4) flash attention
  attn_kernel<<<dim3(SS / 128, HH, BB), 128, 0, stream>>>(Qb, Kb, Vt, Ob);
  // 5) output projection
  outproj_kernel<<<dim3(MM / 256, DD / 64), 256, 0, stream>>>(Ob, WoT, bo, out);
}
